// SEER_ENCODER_STATE_84593675862720
// MI455X (gfx1250) — matrix operand order repaired and checked by reading, not compiled, not run on hardware
//
#include <hip/hip_runtime.h>

typedef __attribute__((ext_vector_type(16))) __bf16 v16bf;
typedef __attribute__((ext_vector_type(8)))  float  v8f;

// ---- packed weight layout (bf16 elements inside d_ws) ----
#define OFF_TW0 0          // 768x128 -> 98304 elems
#define OFF_TW1 98304      // 128x128 -> 16384
#define OFF_TW2 114688
#define OFF_TW3 131072
#define OFF_WQ  147456     // 3 layers x 16384
#define OFF_WK  196608
#define OFF_WV  245760
#define PK_END  294912     // 589,824 bytes
#define TZ_OFF_BYTES (1u << 20)   // task_z f32, 8192*128*4 = 4 MB

// output slice offsets (floats)
#define OUT_PRED   1048576u            // 8192*128
#define OUT_WRIST  6291456u            // + 8192*640
#define OUT_STATE  11534336u           // + 8192*640

// ---------------- WMMA fragment helpers (contiguous-K layouts, k = 16h + s) ----------------

// A-matrix 16x32 bf16 from row-major storage (rows = M, contiguous K).
// lanes 0-15: M=lane, K = kOff+0..15; lanes 16-31: M=lane-16, K = kOff+16..31
__device__ __forceinline__ v16bf load_a_rm(const __bf16* base, int strideElems,
                                           int m_tile, int kOff, int lane) {
  int row = m_tile * 16 + (lane & 15);
  int k0  = kOff + ((lane & 16) ? 16 : 0);
  const uint4* p0 = (const uint4*)(base + row * strideElems + k0);
  const uint4* p1 = (const uint4*)(base + row * strideElems + k0 + 8);
  v16bf a;
  ((uint4*)&a)[0] = *p0;
  ((uint4*)&a)[1] = *p1;
  return a;
}

// B-matrix 32x16 bf16 from N-major storage (row index = N, contiguous K).
// lanes 0-15: N=lane, K = kOff+0..15; lanes 16-31: N=lane-16, K = kOff+16..31
__device__ __forceinline__ v16bf load_b_nmajor(const __bf16* base, int strideElems,
                                               int n_tile, int kOff, int lane) {
  int row = n_tile * 16 + (lane & 15);
  int k0  = kOff + ((lane & 16) ? 16 : 0);
  const uint4* p = (const uint4*)(base + row * strideElems + k0);
  v16bf b;
  ((uint4*)&b)[0] = p[0];
  ((uint4*)&b)[1] = p[1];
  return b;
}

// B fragment from pre-packed storage: 16 contiguous bf16 per lane.
__device__ __forceinline__ v16bf load_b_packed(const __bf16* frag, int lane) {
  const uint4* p = (const uint4*)(frag + lane * 16);
  v16bf b;
  ((uint4*)&b)[0] = p[0];
  ((uint4*)&b)[1] = p[1];
  return b;
}

__device__ __forceinline__ v8f wmma_bf16(v16bf a, v16bf b, v8f c) {
  return __builtin_amdgcn_wmma_f32_16x16x32_bf16(false, a, false, b, (short)0, c,
                                                 false, false);
}

__device__ __forceinline__ float gelu_exact(float x) {
  return 0.5f * x * (1.0f + erff(x * 0.7071067811865475f));
}

// --------------- weight re-pack: f32 (K x N, row-major) -> bf16 fragments ---
__global__ void pack_weight(const float* __restrict__ src, __bf16* __restrict__ dst,
                            int Ktot, int Ntot) {
  int Ksteps = Ktot >> 5;
  int total = Ktot * Ntot;
  for (int idx = blockIdx.x * blockDim.x + threadIdx.x; idx < total;
       idx += gridDim.x * blockDim.x) {
    int e    = idx & 15;
    int lane = (idx >> 4) & 31;
    int fk   = idx >> 9;                 // n_tile*Ksteps + kstep
    int kstep  = fk % Ksteps;
    int n_tile = fk / Ksteps;
    int nn = n_tile * 16 + (lane & 15);
    int kk = kstep * 32 + ((lane & 16) ? 16 : 0) + e;
    dst[idx] = (__bf16)src[kk * Ntot + nn];
  }
}

// --------------------------- task MLP (batched GEMM) ------------------------
// 16 batch rows per block; 4 waves x 2 N-tiles each; bf16 WMMA, f32 accum.
__global__ void __launch_bounds__(128)
task_mlp_kernel(const float* __restrict__ task,
                const float* __restrict__ tb0, const float* __restrict__ tb1,
                const float* __restrict__ tb2, const float* __restrict__ tb3,
                const __bf16* __restrict__ pk, float* __restrict__ task_z) {
  __shared__ __align__(16) __bf16 sh_t[16 * 768];     // staged task rows
  __shared__ __align__(16) __bf16 sh_h[2][16 * 128];  // double-buffered hidden
  const int tid  = threadIdx.x;
  const int lane = tid & 31;
  const int w    = tid >> 5;
  const int b0   = blockIdx.x * 16;

  for (int idx = tid; idx < 16 * 768; idx += 128) {
    int r = idx / 768, c = idx - r * 768;
    sh_t[idx] = (__bf16)task[(size_t)(b0 + r) * 768 + c];
  }
  __syncthreads();

  // layer 0: (16x768) @ (768x128), 24 K-steps
  for (int tt = 0; tt < 2; ++tt) {
    int n_tile = w + tt * 4;
    v8f acc = {};
    for (int ks = 0; ks < 24; ++ks) {
      v16bf a = load_a_rm(sh_t, 768, 0, ks * 32, lane);
      v16bf b = load_b_packed(pk + OFF_TW0 + (n_tile * 24 + ks) * 512, lane);
      acc = wmma_bf16(a, b, acc);
    }
    float bias = tb0[n_tile * 16 + (lane & 15)];
    int col = n_tile * 16 + (lane & 15);
    int row0 = (lane & 16) ? 8 : 0;
#pragma unroll
    for (int r = 0; r < 8; ++r)
      sh_h[0][(row0 + r) * 128 + col] = (__bf16)gelu_exact(acc[r] + bias);
  }
  __syncthreads();

  // layers 1,2: 128x128 + gelu
  for (int L = 0; L < 2; ++L) {
    const __bf16* wb = pk + (L == 0 ? OFF_TW1 : OFF_TW2);
    const float* bp  = (L == 0 ? tb1 : tb2);
    const __bf16* src = sh_h[L & 1];
    __bf16* dst = sh_h[(L + 1) & 1];
    for (int tt = 0; tt < 2; ++tt) {
      int n_tile = w + tt * 4;
      v8f acc = {};
      for (int ks = 0; ks < 4; ++ks) {
        v16bf a = load_a_rm(src, 128, 0, ks * 32, lane);
        v16bf b = load_b_packed(wb + (n_tile * 4 + ks) * 512, lane);
        acc = wmma_bf16(a, b, acc);
      }
      float bias = bp[n_tile * 16 + (lane & 15)];
      int col = n_tile * 16 + (lane & 15);
      int row0 = (lane & 16) ? 8 : 0;
#pragma unroll
      for (int r = 0; r < 8; ++r)
        dst[(row0 + r) * 128 + col] = (__bf16)gelu_exact(acc[r] + bias);
    }
    __syncthreads();
  }

  // layer 3: 128x128, no gelu, f32 out
  for (int tt = 0; tt < 2; ++tt) {
    int n_tile = w + tt * 4;
    v8f acc = {};
    for (int ks = 0; ks < 4; ++ks) {
      v16bf a = load_a_rm(sh_h[0], 128, 0, ks * 32, lane);
      v16bf b = load_b_packed(pk + OFF_TW3 + (n_tile * 4 + ks) * 512, lane);
      acc = wmma_bf16(a, b, acc);
    }
    float bias = tb3[n_tile * 16 + (lane & 15)];
    int col = n_tile * 16 + (lane & 15);
    int row0 = (lane & 16) ? 8 : 0;
#pragma unroll
    for (int r = 0; r < 8; ++r)
      task_z[(size_t)(b0 + row0 + r) * 128 + col] = acc[r] + bias;
  }
}

// ------------------- fused 3-layer attention, one item per block ------------
__global__ void __launch_bounds__(128)
attn_kernel(const float* __restrict__ obs, const float* __restrict__ obsw,
            const float* __restrict__ stt, const float* __restrict__ task_z,
            const float* __restrict__ atok, const float* __restrict__ ptok,
            const float* __restrict__ ln_g, const float* __restrict__ ln_b,
            const float* __restrict__ bq, const float* __restrict__ bk,
            const float* __restrict__ bv, const __bf16* __restrict__ pk,
            const unsigned char* __restrict__ maskp, float* __restrict__ out) {
  __shared__ __align__(16) __bf16 sh_x[32 * 128];   // x / xn (in-place LN)
  __shared__ __align__(16) __bf16 sh_q[32 * 128];   // token-major
  __shared__ __align__(16) __bf16 sh_k[32 * 128];   // token-major
  __shared__ __align__(16) __bf16 sh_vt[128 * 32];  // dim-major (transposed)
  __shared__ __align__(16) __bf16 sh_w[8192];       // staged half-weight
  __shared__ __align__(16) float  sh_s[32 * 32];    // scores f32
  __shared__ __align__(16) __bf16 sh_p[32 * 32];    // probs bf16
  __shared__ unsigned char sh_m[1024];
  __shared__ float sh_sum[128], sh_sq[128], sh_mu[32], sh_rs[32];

  const int tid  = threadIdx.x;
  const int lane = tid & 31;
  const int w    = tid >> 5;
  const size_t b = blockIdx.x;

  // assemble the 32-token sequence in bf16
  for (int idx = tid; idx < 4096; idx += 128) {
    int tk = idx >> 7, d = idx & 127;
    float v;
    if (tk < 5)        v = obs [(b * 5 + tk) * 128 + d];
    else if (tk < 10)  v = obsw[(b * 5 + (tk - 5)) * 128 + d];
    else if (tk < 15)  v = stt [(b * 5 + (tk - 10)) * 128 + d];
    else if (tk == 15) v = task_z[b * 128 + d];
    else if (tk == 16) v = atok[d];
    else               v = ptok[(tk - 17) * 128 + d];
    sh_x[tk * 128 + d] = (__bf16)v;
  }
  for (int j = tid; j < 1024; j += 128) sh_m[j] = maskp[j];
  __syncthreads();

  for (int layer = 0; layer < 3; ++layer) {
    // ---- LayerNorm (in place), 4 threads per token ----
    {
      int tok = tid & 31, part = tid >> 5;
      float s = 0.f, sq = 0.f;
      for (int d = part * 32; d < part * 32 + 32; ++d) {
        float v = (float)sh_x[tok * 128 + d];
        s += v; sq += v * v;
      }
      sh_sum[tok * 4 + part] = s;
      sh_sq [tok * 4 + part] = sq;
      __syncthreads();
      if (tid < 32) {
        float ts = 0.f, tq = 0.f;
        for (int p2 = 0; p2 < 4; ++p2) { ts += sh_sum[tid * 4 + p2]; tq += sh_sq[tid * 4 + p2]; }
        float mu = ts * (1.0f / 128.0f);
        float var = tq * (1.0f / 128.0f) - mu * mu;
        sh_mu[tid] = mu;
        sh_rs[tid] = rsqrtf(var + 1e-5f);
      }
      __syncthreads();
      float mu = sh_mu[tok], rs = sh_rs[tok];
      for (int d = part * 32; d < part * 32 + 32; ++d) {
        float v = (float)sh_x[tok * 128 + d];
        float g = ln_g[layer * 128 + d], be = ln_b[layer * 128 + d];
        sh_x[tok * 128 + d] = (__bf16)((v - mu) * rs * g + be);
      }
      __syncthreads();
    }

    // ---- Q/K/V projections: stage 16KB weight halves, bf16 WMMA ----
    for (int mat = 0; mat < 3; ++mat) {
      const __bf16* wbase = pk + (mat == 0 ? OFF_WQ : mat == 1 ? OFF_WK : OFF_WV)
                               + layer * 16384;
      const float* biasp = (mat == 0 ? bq : mat == 1 ? bk : bv) + layer * 128;
      for (int h = 0; h < 2; ++h) {
        const uint4* src = (const uint4*)(wbase + h * 8192);
        uint4* dw = (uint4*)sh_w;
        for (int j = tid; j < 1024; j += 128) dw[j] = src[j];
        __syncthreads();
        for (int tt = 0; tt < 2; ++tt) {
          int t = w + tt * 4;                // 0..7 tiles this half
          int m_tile = t >> 2;
          int n_sub  = t & 3;
          int n_tile = h * 4 + n_sub;
          v8f acc = {};
          for (int ks = 0; ks < 4; ++ks) {
            v16bf a = load_a_rm(sh_x, 128, m_tile, ks * 32, lane);
            v16bf bf = load_b_packed(sh_w + (n_sub * 4 + ks) * 512, lane);
            acc = wmma_bf16(a, bf, acc);
          }
          float bias = biasp[n_tile * 16 + (lane & 15)];
          int col = n_tile * 16 + (lane & 15);
          int row0 = m_tile * 16 + ((lane & 16) ? 8 : 0);
          if (mat < 2) {
            __bf16* dst = (mat == 0) ? sh_q : sh_k;
#pragma unroll
            for (int r = 0; r < 8; ++r)
              dst[(row0 + r) * 128 + col] = (__bf16)(acc[r] + bias);
          } else {
#pragma unroll
            for (int r = 0; r < 8; ++r)
              sh_vt[col * 32 + row0 + r] = (__bf16)(acc[r] + bias);
          }
        }
        __syncthreads();
      }
    }

    // ---- S = Q @ K^T (32x32), one 16x16 tile per wave ----
    {
      int mq = w >> 1, nk = w & 1;
      v8f acc = {};
      for (int ks = 0; ks < 4; ++ks) {
        v16bf a  = load_a_rm(sh_q, 128, mq, ks * 32, lane);
        v16bf bf = load_b_nmajor(sh_k, 128, nk, ks * 32, lane);
        acc = wmma_bf16(a, bf, acc);
      }
      int col = nk * 16 + (lane & 15);
      int row0 = mq * 16 + ((lane & 16) ? 8 : 0);
#pragma unroll
      for (int r = 0; r < 8; ++r) sh_s[(row0 + r) * 32 + col] = acc[r];
      __syncthreads();
    }

    // ---- masked softmax, one row per lane (wave 0) ----
    if (tid < 32) {
      int row = tid;
      float mx = -3.4e38f;
      float vals[32];
#pragma unroll
      for (int k2 = 0; k2 < 32; ++k2) {
        float v = sh_m[row * 32 + k2] ? -INFINITY : sh_s[row * 32 + k2];
        vals[k2] = v;
        mx = fmaxf(mx, v);
      }
      float den = 0.f;
#pragma unroll
      for (int k2 = 0; k2 < 32; ++k2) { vals[k2] = __expf(vals[k2] - mx); den += vals[k2]; }
      float inv = 1.0f / den;
#pragma unroll
      for (int k2 = 0; k2 < 32; ++k2) sh_p[row * 32 + k2] = (__bf16)(vals[k2] * inv);
    }
    __syncthreads();

    // ---- x = P @ V : K=32 so a single WMMA per 16x16 tile; 4 tiles/wave ----
    for (int tt = 0; tt < 4; ++tt) {
      int t = w + tt * 4;                 // 0..15
      int m_tile = t >> 3;
      int n_tile = t & 7;
      v8f acc = {};
      v16bf a  = load_a_rm(sh_p, 32, m_tile, 0, lane);
      v16bf bf = load_b_nmajor(sh_vt, 32, n_tile, 0, lane);
      acc = wmma_bf16(a, bf, acc);
      int col = n_tile * 16 + (lane & 15);
      int row0 = m_tile * 16 + ((lane & 16) ? 8 : 0);
#pragma unroll
      for (int r = 0; r < 8; ++r)
        sh_x[(row0 + r) * 128 + col] = (__bf16)acc[r];
    }
    __syncthreads();
  }

  // ---- scatter tokens 16..31 into the four output slices (f32) ----
  for (int idx = tid; idx < 2048; idx += 128) {
    int tk = idx >> 7, d = idx & 127;
    float v = (float)sh_x[(16 + tk) * 128 + d];
    size_t dst;
    if (tk == 0)      dst = b * 128 + d;
    else if (tk <= 5) dst = OUT_PRED  + (b * 5 + (tk - 1))  * 128 + d;
    else if (tk <= 10)dst = OUT_WRIST + (b * 5 + (tk - 6))  * 128 + d;
    else              dst = OUT_STATE + (b * 5 + (tk - 11)) * 128 + d;
    out[dst] = v;
  }
}

// ------------------------------- launcher -----------------------------------
extern "C" void kernel_launch(void* const* d_in, const int* in_sizes, int n_in,
                              void* d_out, int out_size, void* d_ws, size_t ws_size,
                              hipStream_t stream) {
  const float* obs  = (const float*)d_in[0];
  const float* obsw = (const float*)d_in[1];
  const float* stt  = (const float*)d_in[2];
  const float* task = (const float*)d_in[3];
  const float* tw0  = (const float*)d_in[4];
  const float* tb0  = (const float*)d_in[5];
  const float* tw1  = (const float*)d_in[6];
  const float* tb1  = (const float*)d_in[7];
  const float* tw2  = (const float*)d_in[8];
  const float* tb2  = (const float*)d_in[9];
  const float* tw3  = (const float*)d_in[10];
  const float* tb3  = (const float*)d_in[11];
  const float* atok = (const float*)d_in[12];
  const float* ptok = (const float*)d_in[13];
  const float* lng  = (const float*)d_in[14];
  const float* lnb  = (const float*)d_in[15];
  const float* wq   = (const float*)d_in[16];
  const float* bq   = (const float*)d_in[17];
  const float* wk   = (const float*)d_in[18];
  const float* bk   = (const float*)d_in[19];
  const float* wv   = (const float*)d_in[20];
  const float* bv   = (const float*)d_in[21];
  const unsigned char* mask = (const unsigned char*)d_in[22];

  __bf16* pk     = (__bf16*)d_ws;
  float*  task_z = (float*)((char*)d_ws + TZ_OFF_BYTES);
  float*  out    = (float*)d_out;

  // re-pack all GEMM weights into bf16 B-fragment order (L2-resident, 590 KB)
  pack_weight<<<96, 256, 0, stream>>>(tw0, pk + OFF_TW0, 768, 128);
  pack_weight<<<16, 256, 0, stream>>>(tw1, pk + OFF_TW1, 128, 128);
  pack_weight<<<16, 256, 0, stream>>>(tw2, pk + OFF_TW2, 128, 128);
  pack_weight<<<16, 256, 0, stream>>>(tw3, pk + OFF_TW3, 128, 128);
  for (int i = 0; i < 3; ++i) {
    pack_weight<<<16, 256, 0, stream>>>(wq + i * 16384, pk + OFF_WQ + i * 16384, 128, 128);
    pack_weight<<<16, 256, 0, stream>>>(wk + i * 16384, pk + OFF_WK + i * 16384, 128, 128);
    pack_weight<<<16, 256, 0, stream>>>(wv + i * 16384, pk + OFF_WV + i * 16384, 128, 128);
  }

  // task MLP: 8192 rows / 16 rows-per-block
  task_mlp_kernel<<<512, 128, 0, stream>>>(task, tb0, tb1, tb2, tb3, pk, task_z);

  // fused 3-layer attention, one batch item per 4-wave block
  attn_kernel<<<8192, 128, 0, stream>>>(obs, obsw, stt, task_z, atok, ptok,
                                        lng, lnb, bq, bk, bv, pk, mask, out);
  (void)in_sizes; (void)n_in; (void)out_size; (void)ws_size;
}
